// MultitaskEncoder_23493471109267
// MI455X (gfx1250) — compile-verified, matrix operand order checked by reading, not run on hardware
//
#include <hip/hip_runtime.h>
#include <hip/hip_bf16.h>

// ---------------- model constants ----------------
constexpr int kB  = 2;
constexpr int kT  = 1024;
constexpr int kD  = 512;
constexpr int kNH = 8;
constexpr int kHD = 64;     // kD / kNH
constexpr int kL  = 4;
constexpr int kH  = 2048;   // 4*kD
constexpr int kE  = 8;
constexpr int kNT = kB * kT;          // 2048 tokens
constexpr float kEPS = 1e-5f;

// ---------------- WMMA types ----------------
typedef __attribute__((ext_vector_type(16))) __bf16 v16bf;
typedef __attribute__((ext_vector_type(8)))  float  v8f;
typedef __attribute__((ext_vector_type(4)))  int    v4i;

union AFrag {
    uint4  u[2];
    v16bf  v;
    __bf16 e[16];
};

static __device__ __forceinline__ v8f wmma_bf16(v16bf a, v16bf b, v8f c) {
    return __builtin_amdgcn_wmma_f32_16x16x32_bf16(
        /*neg_a=*/false, a, /*neg_b=*/false, b,
        /*c_mod=*/(short)0, c, /*reuse_a=*/false, /*reuse_b=*/false);
}

// ---------------- gfx1250 async global->LDS copy (16B per lane) ----------------
#if defined(__has_builtin)
#if __has_builtin(__builtin_amdgcn_global_load_async_to_lds_b128)
#define USE_ASYNC_LDS 1
#endif
#endif

static __device__ __forceinline__ void cp16(void* lds_dst, const void* gsrc) {
#ifdef USE_ASYNC_LDS
    typedef __attribute__((address_space(1))) v4i GV;
    typedef __attribute__((address_space(3))) v4i LV;
    __builtin_amdgcn_global_load_async_to_lds_b128(
        (GV*)gsrc, (LV*)lds_dst, 0, 0);
#else
    *(uint4*)lds_dst = *(const uint4*)gsrc;
#endif
}

static __device__ __forceinline__ void wait_async() {
#ifdef USE_ASYNC_LDS
#if __has_builtin(__builtin_amdgcn_s_wait_asynccnt)
    __builtin_amdgcn_s_wait_asynccnt(0);
#else
    asm volatile("s_wait_asynccnt 0" ::: "memory");
#endif
#endif
}

// =====================================================================
// Generic GEMM:  C[M,N] (+)= A[M,K] * W[N,K]^T   (bf16 in, fp32 accum)
// Block tile 128x64, K-step 32, 256 threads = 8 waves (4x2), each wave
// a 32x32 tile as 2x2 WMMA.  Tiles staged via double-buffered async
// global->LDS b128 copies (no conversion: operands are already bf16).
// MODE: 0 = store fp32, 1 = C += res (fp32), 2 = dual-GEMM SwiGLU -> bf16
//       3 = C += rowscale[m*rss] * res  (MoE combine, fp32)
// =====================================================================
constexpr int GBM = 128, GBN = 64, GBK = 32;
constexpr int GLD = 56;   // LDS row stride in bf16: 112B (16B aligned, conflict-friendly)

template<int MODE>
__global__ __launch_bounds__(256)
void gemm_wmma(const __bf16* __restrict__ A, const __bf16* __restrict__ W,
               const __bf16* __restrict__ W3, void* __restrict__ Cv,
               const float* __restrict__ rowscale, int rss,
               int M, int N, int K)
{
    __shared__ __bf16 As[2][GBM][GLD];
    __shared__ __bf16 Ws[2][GBN][GLD];
    __shared__ __bf16 W3s[2][(MODE == 2) ? GBN : 1][(MODE == 2) ? GLD : 1];

    const int tid  = threadIdx.x;
    const int lane = tid & 31;
    const int wv   = tid >> 5;
    const int wm   = wv & 3;
    const int wn   = wv >> 2;
    const int bM   = blockIdx.y * GBM;
    const int bN   = blockIdx.x * GBN;

    const int mlo    = lane & 15;
    const int kb     = (lane < 16) ? 0 : 8;
    const int rowoff = (lane < 16) ? 0 : 8;

    v8f acc[2][2]  = {};
    v8f acc3[2][2] = {};

    // stage tile for K-chunk s into buffer b (async b128 per lane)
    auto stage = [&](int s, int b) {
        const int k0 = s * GBK;
        #pragma unroll
        for (int it = 0; it < 2; ++it) {                 // A: 128 rows x 4 chunks
            const int idx = tid + it * 256;
            const int row = idx >> 2, q = (idx & 3) * 8;
            cp16(&As[b][row][q], A + (size_t)(bM + row) * K + k0 + q);
        }
        {                                                // W: 64 rows x 4 chunks
            const int row = tid >> 2, q = (tid & 3) * 8;
            cp16(&Ws[b][row][q], W + (size_t)(bN + row) * K + k0 + q);
            if (MODE == 2)
                cp16(&W3s[b][row][q], W3 + (size_t)(bN + row) * K + k0 + q);
        }
    };

    const int nk = K / GBK;
    int buf = 0;
    stage(0, 0);
    for (int s = 0; s < nk; ++s) {
        wait_async();
        __syncthreads();
        if (s + 1 < nk) stage(s + 1, buf ^ 1);

        AFrag af[2], bf[2], bf3[2];
        #pragma unroll
        for (int i = 0; i < 2; ++i) {
            const int m = wm * 32 + i * 16 + mlo;
            af[i].u[0] = *(const uint4*)&As[buf][m][kb];
            af[i].u[1] = *(const uint4*)&As[buf][m][kb + 16];
            const int n = wn * 32 + i * 16 + mlo;
            bf[i].u[0] = *(const uint4*)&Ws[buf][n][kb];
            bf[i].u[1] = *(const uint4*)&Ws[buf][n][kb + 16];
            if (MODE == 2) {
                bf3[i].u[0] = *(const uint4*)&W3s[buf][n][kb];
                bf3[i].u[1] = *(const uint4*)&W3s[buf][n][kb + 16];
            }
        }
        #pragma unroll
        for (int i = 0; i < 2; ++i)
            #pragma unroll
            for (int j = 0; j < 2; ++j) {
                acc[i][j] = wmma_bf16(af[i].v, bf[j].v, acc[i][j]);
                if (MODE == 2) acc3[i][j] = wmma_bf16(af[i].v, bf3[j].v, acc3[i][j]);
            }
        buf ^= 1;
    }

    // ---- epilogue ----
    #pragma unroll
    for (int i = 0; i < 2; ++i)
        #pragma unroll
        for (int j = 0; j < 2; ++j) {
            const int m0 = bM + wm * 32 + i * 16 + rowoff;
            const int n  = bN + wn * 32 + j * 16 + mlo;
            #pragma unroll
            for (int r = 0; r < 8; ++r) {
                const int m = m0 + r;
                const float val = acc[i][j][r];
                if (MODE == 0) {
                    ((float*)Cv)[(size_t)m * N + n] = val;
                } else if (MODE == 1) {
                    ((float*)Cv)[(size_t)m * N + n] += val;
                } else if (MODE == 2) {
                    const float u = acc3[i][j][r];
                    ((__bf16*)Cv)[(size_t)m * N + n] =
                        (__bf16)((val / (1.f + __expf(-val))) * u);   // silu(g)*u
                } else {
                    ((float*)Cv)[(size_t)m * N + n] +=
                        rowscale[(size_t)m * rss] * val;
                }
            }
        }
}

// =====================================================================
// Flash attention (causal).  Q,K : [B*NH][T][HD] bf16 (Q pre-scaled by
// 1/sqrt(HD)); V : [B*NH][HD][T] bf16 (transposed).  O : [B*T][D] bf16.
// 4 waves / block, 16 q rows per wave; double-buffered async K/V tiles.
// =====================================================================
__global__ __launch_bounds__(128)
void flash_attn(const __bf16* __restrict__ Q, const __bf16* __restrict__ K,
                const __bf16* __restrict__ V, __bf16* __restrict__ O)
{
    __shared__ __bf16 Ks[2][32][72];      // [krow][hd]
    __shared__ __bf16 Vt[2][64][72];      // [hd][krow]
    __shared__ __bf16 Ps[4][16][72];      // wave-private P tiles

    const int tid  = threadIdx.x;
    const int lane = tid & 31;
    const int wv   = tid >> 5;
    const int bh   = blockIdx.y;
    const int qblk = blockIdx.x;
    const int qbase = qblk * 64 + wv * 16;

    const __bf16* Kb = K + (size_t)bh * kT * kHD;
    const __bf16* Vb = V + (size_t)bh * kT * kHD;   // [HD][T]

    const int mlo    = lane & 15;
    const int kb     = (lane < 16) ? 0 : 8;
    const int rowoff = (lane < 16) ? 0 : 8;

    // Q fragments: direct bf16 b128 global loads (layout matches A-frag)
    AFrag qf[2];
    {
        const __bf16* qr = Q + ((size_t)bh * kT + qbase + mlo) * kHD;
        #pragma unroll
        for (int c = 0; c < 2; ++c) {
            qf[c].u[0] = *(const uint4*)(qr + c * 32 + kb);
            qf[c].u[1] = *(const uint4*)(qr + c * 32 + kb + 16);
        }
    }

    v8f o[4] = {};
    float mcur[8], lsum[8];
    #pragma unroll
    for (int r = 0; r < 8; ++r) { mcur[r] = -1e30f; lsum[r] = 0.f; }

    auto stageKV = [&](int j, int b) {
        const int kt = j * 32;
        #pragma unroll
        for (int it = 0; it < 2; ++it) {
            const int idx = tid + it * 128;
            {   // K: 32 rows x 8 chunks of 16B
                const int kr = idx >> 3, ch = (idx & 7) * 8;
                cp16(&Ks[b][kr][ch], Kb + (size_t)(kt + kr) * kHD + ch);
            }
            {   // V^T: 64 rows x 4 chunks of 16B
                const int hd = idx >> 2, ch = (idx & 3) * 8;
                cp16(&Vt[b][hd][ch], Vb + (size_t)hd * kT + kt + ch);
            }
        }
    };

    const int jend = qblk * 2 + 1;
    int buf = 0;
    stageKV(0, 0);
    for (int j = 0; j <= jend; ++j) {
        const int kt = j * 32;
        wait_async();
        __syncthreads();
        if (j + 1 <= jend) stageKV(j + 1, buf ^ 1);

        if (kt <= qbase + 15) {                      // wave-uniform
            v8f s[2];
            #pragma unroll
            for (int jn = 0; jn < 2; ++jn) {
                const int n = jn * 16 + mlo;
                AFrag b0, b1;
                b0.u[0] = *(const uint4*)&Ks[buf][n][kb];
                b0.u[1] = *(const uint4*)&Ks[buf][n][kb + 16];
                b1.u[0] = *(const uint4*)&Ks[buf][n][32 + kb];
                b1.u[1] = *(const uint4*)&Ks[buf][n][32 + kb + 16];
                v8f z = {};
                z = wmma_bf16(qf[0].v, b0.v, z);
                z = wmma_bf16(qf[1].v, b1.v, z);
                s[jn] = z;
            }
            float sc[8];
            #pragma unroll
            for (int r = 0; r < 8; ++r) {
                const int qg = qbase + r + rowoff;
                float v0 = s[0][r]; if (kt + mlo      > qg) v0 = -1e30f;
                float v1 = s[1][r]; if (kt + 16 + mlo > qg) v1 = -1e30f;
                float mx = fmaxf(v0, v1);
                #pragma unroll
                for (int off = 8; off >= 1; off >>= 1)
                    mx = fmaxf(mx, __shfl_xor(mx, off));
                const float nm  = fmaxf(mcur[r], mx);
                const float scl = __expf(mcur[r] - nm);
                mcur[r] = nm;
                const float p0 = __expf(v0 - nm);
                const float p1 = __expf(v1 - nm);
                s[0][r] = p0; s[1][r] = p1;
                float ps = p0 + p1;
                #pragma unroll
                for (int off = 8; off >= 1; off >>= 1)
                    ps += __shfl_xor(ps, off);
                lsum[r] = lsum[r] * scl + ps;
                sc[r] = scl;
            }
            #pragma unroll
            for (int t = 0; t < 4; ++t)
                #pragma unroll
                for (int r = 0; r < 8; ++r) o[t][r] *= sc[r];

            #pragma unroll
            for (int r = 0; r < 8; ++r) {
                const int m = r + rowoff;
                Ps[wv][m][mlo]      = (__bf16)s[0][r];
                Ps[wv][m][16 + mlo] = (__bf16)s[1][r];
            }
            asm volatile("s_wait_dscnt 0" ::: "memory");
            AFrag pf;
            pf.u[0] = *(const uint4*)&Ps[wv][mlo][kb];
            pf.u[1] = *(const uint4*)&Ps[wv][mlo][kb + 16];

            #pragma unroll
            for (int t = 0; t < 4; ++t) {
                const int hd = t * 16 + mlo;
                AFrag vfr;
                vfr.u[0] = *(const uint4*)&Vt[buf][hd][kb];
                vfr.u[1] = *(const uint4*)&Vt[buf][hd][kb + 16];
                o[t] = wmma_bf16(pf.v, vfr.v, o[t]);
            }
        }
        buf ^= 1;
    }

    const int b = bh / kNH, hh = bh % kNH;
    #pragma unroll
    for (int t = 0; t < 4; ++t)
        #pragma unroll
        for (int r = 0; r < 8; ++r) {
            const int qg  = qbase + r + rowoff;
            const int col = hh * kHD + t * 16 + mlo;
            O[(size_t)(b * kT + qg) * kD + col] = (__bf16)(o[t][r] / lsum[r]);
        }
}

// =====================================================================
// RMSNorm: fp32 in, fp32 or bf16 out. One 128-thread block per row.
// =====================================================================
template<bool BF16OUT>
__global__ __launch_bounds__(128)
void rmsnorm_k(const float* __restrict__ x, const float* __restrict__ w,
               void* __restrict__ y)
{
    const int row = blockIdx.x;
    const float* xr = x + (size_t)row * kD;
    const float4 v = *(const float4*)(xr + threadIdx.x * 4);
    float ss = v.x * v.x + v.y * v.y + v.z * v.z + v.w * v.w;
    #pragma unroll
    for (int off = 16; off >= 1; off >>= 1) ss += __shfl_xor(ss, off);
    __shared__ float red[4];
    const int lane = threadIdx.x & 31, wv = threadIdx.x >> 5;
    if (lane == 0) red[wv] = ss;
    __syncthreads();
    const float tot = red[0] + red[1] + red[2] + red[3];
    const float rs = rsqrtf(tot / (float)kD + kEPS);
    const float4 wv4 = *(const float4*)(w + threadIdx.x * 4);
    if (BF16OUT) {
        __bf16 o4[4] = { (__bf16)(v.x * wv4.x * rs), (__bf16)(v.y * wv4.y * rs),
                         (__bf16)(v.z * wv4.z * rs), (__bf16)(v.w * wv4.w * rs) };
        *(uint2*)((__bf16*)y + (size_t)row * kD + threadIdx.x * 4) = *(uint2*)o4;
    } else {
        float4 oy;
        oy.x = v.x * wv4.x * rs; oy.y = v.y * wv4.y * rs;
        oy.z = v.z * wv4.z * rs; oy.w = v.w * wv4.w * rs;
        *(float4*)((float*)y + (size_t)row * kD + threadIdx.x * 4) = oy;
    }
}

// =====================================================================
// RoPE + split qkv[B*T,3D] fp32 -> bf16 Q (scaled), K : [B*NH][T][HD],
// V transposed : [B*NH][HD][T]
// =====================================================================
__global__ __launch_bounds__(256)
void rope_split(const float* __restrict__ qkv, __bf16* __restrict__ Qo,
                __bf16* __restrict__ Ko, __bf16* __restrict__ Vo)
{
    const int idx = blockIdx.x * blockDim.x + threadIdx.x;   // B*T*NH*32
    const int d2 = idx & 31;
    const int h  = (idx >> 5) & (kNH - 1);
    const int bt = idx >> 8;
    const int t  = bt & (kT - 1);
    const int b  = bt >> 10;
    const float* src = qkv + (size_t)bt * (3 * kD);
    const float inv = __powf(10000.f, -(float)d2 / 32.f);
    float sn, cs;
    __sincosf((float)t * inv, &sn, &cs);
    const int bhh = b * kNH + h;
    const size_t dst = ((size_t)bhh * kT + t) * kHD;
    const float q1 = src[h * kHD + d2], q2 = src[h * kHD + d2 + 32];
    Qo[dst + d2]      = (__bf16)((q1 * cs - q2 * sn) * 0.125f);   // fold 1/sqrt(HD)
    Qo[dst + d2 + 32] = (__bf16)((q1 * sn + q2 * cs) * 0.125f);
    const float k1 = src[kD + h * kHD + d2], k2 = src[kD + h * kHD + d2 + 32];
    Ko[dst + d2]      = (__bf16)(k1 * cs - k2 * sn);
    Ko[dst + d2 + 32] = (__bf16)(k1 * sn + k2 * cs);
    const size_t vdst = ((size_t)bhh * kHD);
    Vo[(vdst + d2)      * kT + t] = (__bf16)src[2 * kD + h * kHD + d2];
    Vo[(vdst + d2 + 32) * kT + t] = (__bf16)src[2 * kD + h * kHD + d2 + 32];
}

// =====================================================================
// fp32 -> bf16 weight conversion (vectorized, n multiple of 4)
// =====================================================================
__global__ __launch_bounds__(256)
void cvt_bf16(const float* __restrict__ s, __bf16* __restrict__ d, int n4)
{
    const int i = blockIdx.x * 256 + threadIdx.x;
    if (i >= n4) return;
    const float4 f = ((const float4*)s)[i];
    __bf16 o4[4] = { (__bf16)f.x, (__bf16)f.y, (__bf16)f.z, (__bf16)f.w };
    *(uint2*)(d + (size_t)i * 4) = *(uint2*)o4;
}

// =====================================================================
// Router logits (xn bf16, router weights fp32): one wave per token
// =====================================================================
__global__ __launch_bounds__(256)
void router_logits(const __bf16* __restrict__ xn, const float* __restrict__ rw,
                   float* __restrict__ logits)
{
    const int tok  = blockIdx.x * 8 + (threadIdx.x >> 5);
    const int lane = threadIdx.x & 31;
    const __bf16* xr = xn + (size_t)tok * kD;
    for (int e = 0; e < kE; ++e) {
        const float* we = rw + (size_t)e * kD;
        float s = 0.f;
        for (int i = lane; i < kD; i += 32) s += (float)xr[i] * we[i];
        #pragma unroll
        for (int off = 16; off >= 1; off >>= 1) s += __shfl_xor(s, off);
        if (lane == 0) logits[(size_t)tok * kE + e] = s;
    }
}

// top-2 + softmax -> dense combine [N, E]
__global__ __launch_bounds__(256)
void topk_combine(const float* __restrict__ logits, float* __restrict__ comb)
{
    const int t = blockIdx.x * blockDim.x + threadIdx.x;
    if (t >= kNT) return;
    float v[kE];
    #pragma unroll
    for (int e = 0; e < kE; ++e) v[e] = logits[(size_t)t * kE + e];
    int i1 = 0;
    #pragma unroll
    for (int e = 1; e < kE; ++e) if (v[e] > v[i1]) i1 = e;
    int i2 = (i1 == 0) ? 1 : 0;
    #pragma unroll
    for (int e = 0; e < kE; ++e) if (e != i1 && v[e] > v[i2]) i2 = e;
    const float e2 = __expf(v[i2] - v[i1]);
    const float den = 1.f + e2;
    #pragma unroll
    for (int e = 0; e < kE; ++e) comb[(size_t)t * kE + e] = 0.f;
    comb[(size_t)t * kE + i1] = 1.f / den;
    comb[(size_t)t * kE + i2] = e2 / den;
}

// =====================================================================
// Orchestration
// =====================================================================
extern "C" void kernel_launch(void* const* d_in, const int* in_sizes, int n_in,
                              void* d_out, int out_size, void* d_ws, size_t ws_size,
                              hipStream_t stream)
{
    (void)in_sizes; (void)n_in; (void)out_size; (void)ws_size;

    const float* x     = (const float*)d_in[0];
    const float* qkv_w = (const float*)d_in[1];
    const float* out_w = (const float*)d_in[2];
    const float* n1_w  = (const float*)d_in[3];
    const float* n2_w  = (const float*)d_in[4];
    const float* rt_w  = (const float*)d_in[5];
    const float* mw1   = (const float*)d_in[6];
    const float* mw2   = (const float*)d_in[7];
    const float* mw3   = (const float*)d_in[8];
    const float* fw1   = (const float*)d_in[9];
    const float* fw2   = (const float*)d_in[10];
    const float* fw3   = (const float*)d_in[11];
    const float* nf_w  = (const float*)d_in[12];

    // ---- fp32 workspace region ----
    float* wsf  = (float*)d_ws;
    float* h    = wsf;                              // [NT, D]
    float* qkvf = h    + (size_t)kNT * kD;          // [NT, 3D]
    float* lg   = qkvf + (size_t)kNT * 3 * kD;      // [NT, E]
    float* comb = lg   + (size_t)kNT * kE;          // [NT, E]
    // ---- bf16 workspace region ----
    __bf16* wsb = (__bf16*)(comb + (size_t)kNT * kE);
    __bf16* xnb  = wsb;                             // [NT, D]
    __bf16* Qb   = xnb  + (size_t)kNT * kD;         // [B*NH][T][HD]
    __bf16* Kb   = Qb   + (size_t)kNT * kD;         // [B*NH][T][HD]
    __bf16* Vtb  = Kb   + (size_t)kNT * kD;         // [B*NH][HD][T]
    __bf16* aob  = Vtb  + (size_t)kNT * kD;         // [NT, D]
    __bf16* ffhb = aob  + (size_t)kNT * kD;         // [NT, H]
    // bf16 weights
    __bf16* bqkv = ffhb + (size_t)kNT * kH;
    __bf16* bout = bqkv + (size_t)kL * 3 * kD * kD;
    __bf16* bmw1 = bout + (size_t)kL * kD * kD;
    __bf16* bmw2 = bmw1 + (size_t)2 * kE * kH * kD;
    __bf16* bmw3 = bmw2 + (size_t)2 * kE * kD * kH;
    __bf16* bfw1 = bmw3 + (size_t)2 * kE * kH * kD;
    __bf16* bfw2 = bfw1 + (size_t)2 * kH * kD;
    __bf16* bfw3 = bfw2 + (size_t)2 * kD * kH;

    auto cvt = [&](const float* s, __bf16* d, size_t n) {
        const int n4 = (int)(n / 4);
        cvt_bf16<<<(n4 + 255) / 256, 256, 0, stream>>>(s, d, n4);
    };
    cvt(qkv_w, bqkv, (size_t)kL * 3 * kD * kD);
    cvt(out_w, bout, (size_t)kL * kD * kD);
    cvt(mw1,   bmw1, (size_t)2 * kE * kH * kD);
    cvt(mw2,   bmw2, (size_t)2 * kE * kD * kH);
    cvt(mw3,   bmw3, (size_t)2 * kE * kH * kD);
    cvt(fw1,   bfw1, (size_t)2 * kH * kD);
    cvt(fw2,   bfw2, (size_t)2 * kD * kH);
    cvt(fw3,   bfw3, (size_t)2 * kH * kD);

    (void)hipMemcpyAsync(h, x, (size_t)kNT * kD * sizeof(float),
                         hipMemcpyDeviceToDevice, stream);

    const dim3 blk256(256), blk128(128);
    const dim3 gQKV(3 * kD / GBN, kNT / GBM);
    const dim3 gOUT(kD / GBN,     kNT / GBM);
    const dim3 gFF1(kH / GBN,     kNT / GBM);
    const dim3 gAttn(kT / 64, kB * kNH);

    for (int l = 0; l < kL; ++l) {
        // ---- attention ----
        rmsnorm_k<true><<<kNT, blk128, 0, stream>>>(h, n1_w + (size_t)l * kD, xnb);
        gemm_wmma<0><<<gQKV, blk256, 0, stream>>>(
            xnb, bqkv + (size_t)l * 3 * kD * kD, nullptr, qkvf, nullptr, 0,
            kNT, 3 * kD, kD);
        rope_split<<<(kB * kT * kNH * 32) / 256, blk256, 0, stream>>>(qkvf, Qb, Kb, Vtb);
        flash_attn<<<gAttn, blk128, 0, stream>>>(Qb, Kb, Vtb, aob);
        gemm_wmma<1><<<gOUT, blk256, 0, stream>>>(
            aob, bout + (size_t)l * kD * kD, nullptr, h, nullptr, 0,
            kNT, kD, kD);

        // ---- FFN ----
        rmsnorm_k<true><<<kNT, blk128, 0, stream>>>(h, n2_w + (size_t)l * kD, xnb);
        if ((l & 1) == 0) {
            const int m = l / 2;
            router_logits<<<kNT / 8, blk256, 0, stream>>>(
                xnb, rt_w + (size_t)m * kE * kD, lg);
            topk_combine<<<kNT / 256, blk256, 0, stream>>>(lg, comb);
            for (int e = 0; e < kE; ++e) {
                const size_t wo = ((size_t)m * kE + e) * (size_t)kH * kD;
                gemm_wmma<2><<<gFF1, blk256, 0, stream>>>(
                    xnb, bmw1 + wo, bmw3 + wo, ffhb, nullptr, 0, kNT, kH, kD);
                gemm_wmma<3><<<gOUT, blk256, 0, stream>>>(
                    ffhb, bmw2 + ((size_t)m * kE + e) * (size_t)kD * kH, nullptr, h,
                    comb + e, kE, kNT, kD, kH);
            }
        } else {
            const int m = (l - 1) / 2;
            gemm_wmma<2><<<gFF1, blk256, 0, stream>>>(
                xnb, bfw1 + (size_t)m * kH * kD, bfw3 + (size_t)m * kH * kD, ffhb,
                nullptr, 0, kNT, kH, kD);
            gemm_wmma<1><<<gOUT, blk256, 0, stream>>>(
                ffhb, bfw2 + (size_t)m * kD * kH, nullptr, h, nullptr, 0,
                kNT, kD, kH);
        }
    }

    rmsnorm_k<false><<<kNT, blk128, 0, stream>>>(h, nf_w, (float*)d_out);
}
